// UnityCollapseNetwork_9517647528051
// MI455X (gfx1250) — matrix operand order repaired and checked by reading, not compiled, not run on hardware
//
#include <hip/hip_runtime.h>
#include <hip/hip_bf16.h>
#include <math.h>

// ---------------------------------------------------------------------------
// UnityCollapseNetwork: 3-layer MLP, bf16 WMMA + async-LDS path for gfx1250
//   h1 = sigmoid(PHI*(x @ W1^T + b1))
//   h2 = sigmoid(PHI*(h1 @ W2^T + b2))
//   out = h2 @ W3^T + b3
// ---------------------------------------------------------------------------

typedef __bf16 bf16_t;
typedef __attribute__((ext_vector_type(16))) __bf16 v16bf;
typedef __attribute__((ext_vector_type(8)))  float  v8f;

typedef int v4i __attribute__((vector_size(16)));
typedef __attribute__((address_space(1))) v4i g_v4i;   // global (AS1)
typedef __attribute__((address_space(3))) v4i l_v4i;   // LDS (AS3)
typedef __attribute__((address_space(3))) void lvoid_t;

#define PHI_F 1.61803398874989484820f

#define B_DIM  16384
#define DIN    2048
#define DH     4096

// ---- async global -> LDS copy of 16 bytes per lane (ASYNCcnt-tracked) ----
__device__ __forceinline__ void async_copy_b128(const void* g, void* l) {
#if __has_builtin(__builtin_amdgcn_global_load_async_to_lds_b128)
    __builtin_amdgcn_global_load_async_to_lds_b128(
        (g_v4i*)g, (l_v4i*)l, /*imm_offset=*/0, /*cpol=*/0);
#else
    unsigned lofs = (unsigned)(unsigned long long)(lvoid_t*)l;  // 32-bit LDS addr
    unsigned long long ga = (unsigned long long)g;
    asm volatile("global_load_async_to_lds_b128 %0, %1, off"
                 :: "v"(lofs), "v"(ga) : "memory");
#endif
}

__device__ __forceinline__ void wait_asynccnt0() {
#if __has_builtin(__builtin_amdgcn_s_wait_asynccnt)
    __builtin_amdgcn_s_wait_asynccnt(0);
#else
    asm volatile("s_wait_asynccnt 0x0" ::: "memory");
#endif
}

// ---------------------------------------------------------------------------
// fp32 -> bf16 conversion (vectorized, grid-stride). n4 = element_count / 4.
// ---------------------------------------------------------------------------
__global__ __launch_bounds__(256)
void cvt_f32_to_bf16(const float* __restrict__ in, bf16_t* __restrict__ out, long n4) {
    long stride = (long)gridDim.x * blockDim.x;
    for (long idx = (long)blockIdx.x * blockDim.x + threadIdx.x; idx < n4; idx += stride) {
        long j = idx * 4;
        float4 v = *reinterpret_cast<const float4*>(in + j);
        bf16_t r[4];
        r[0] = (bf16_t)v.x; r[1] = (bf16_t)v.y; r[2] = (bf16_t)v.z; r[3] = (bf16_t)v.w;
        *reinterpret_cast<uint2*>(out + j) = *reinterpret_cast<const uint2*>(r);
    }
}

// ---------------------------------------------------------------------------
// Fused GEMM + bias + sigmoid(PHI*x):  out[M,N](bf16) = act(A[M,K] @ W[N,K]^T + bias)
// Block: 256 threads = 8 waves. Block tile 64(M) x 256(N), K-step 32.
// Wave grid inside block: 2 (M) x 4 (N); each wave computes 32x64 via 8 WMMAs.
// Double-buffered LDS; staging uses global_load_async_to_lds_b128 so the next
// tile streams into LDS (no VGPR round-trip, no per-chunk loadcnt waits) while
// the current tile's WMMAs execute. Drained by one s_wait_asynccnt + barrier.
// ---------------------------------------------------------------------------
__global__ __launch_bounds__(256)
void mlp_gemm_bf16(const bf16_t* __restrict__ A, const bf16_t* __restrict__ W,
                   const float* __restrict__ bias, bf16_t* __restrict__ out,
                   int M, int N, int K) {
    __shared__ bf16_t As[2][64][32];     //  8 KB
    __shared__ bf16_t Ws[2][256][32];    // 32 KB

    const int tid  = threadIdx.x;
    const int lane = tid & 31;
    const int wave = tid >> 5;
    const int wm   = (wave & 1) * 32;   // wave M offset within block tile
    const int wn   = (wave >> 1) * 64;  // wave N offset within block tile

    const int m0 = blockIdx.y * 64;
    const int n0 = blockIdx.x * 256;

    const int h = lane >> 4;            // lane half (0: lanes 0-15, 1: lanes 16-31)
    const int r = lane & 15;

    // cooperative-load geometry: 4 threads per 32-elem (64B) row, 16B per thread
    const int ldr = tid >> 2;           // 0..63
    const int ldc = (tid & 3) * 8;      // 0,8,16,24 (bf16 elems)

    // stage A tile 64x32 and W tile 256x32 into LDS buffer `buf` (async)
    auto stage_tiles = [&](int k0, int buf) {
        async_copy_b128(&A[(size_t)(m0 + ldr) * K + k0 + ldc], &As[buf][ldr][ldc]);
#pragma unroll
        for (int p = 0; p < 4; ++p)
            async_copy_b128(&W[(size_t)(n0 + ldr + 64 * p) * K + k0 + ldc],
                            &Ws[buf][ldr + 64 * p][ldc]);
    };

    v8f c[2][4];
#pragma unroll
    for (int mi = 0; mi < 2; ++mi)
#pragma unroll
        for (int ni = 0; ni < 4; ++ni)
            c[mi][ni] = (v8f){};

    const int nk = K >> 5;  // K/32 steps
    stage_tiles(0, 0);
    wait_asynccnt0();
    __syncthreads();

    for (int kt = 0; kt < nk; ++kt) {
        const int cur = kt & 1;
        if (kt + 1 < nk) stage_tiles((kt + 1) << 5, cur ^ 1);  // overlaps WMMAs below

        // ---- build fragments per ISA 7.12.2 bf16 layouts ----
        // A 16x32: lane half h, row r: elem j<8 -> K = h*8+j ; j>=8 -> K = 16+h*8+(j-8)
        v16bf a[2], b[4];
#pragma unroll
        for (int mi = 0; mi < 2; ++mi) {
#pragma unroll
            for (int j = 0; j < 8; ++j) {
                a[mi][j]     = As[cur][wm + 16 * mi + r][h * 8 + j];
                a[mi][j + 8] = As[cur][wm + 16 * mi + r][16 + h * 8 + j];
            }
        }
        // B 32x16: lane = column r (per half), elem j<8 -> K = h*8+j ; j>=8 -> K = 16+h*8+(j-8)
#pragma unroll
        for (int ni = 0; ni < 4; ++ni) {
#pragma unroll
            for (int j = 0; j < 8; ++j) {
                b[ni][j]     = Ws[cur][wn + 16 * ni + r][h * 8 + j];
                b[ni][j + 8] = Ws[cur][wn + 16 * ni + r][16 + h * 8 + j];
            }
        }

        // ---- 8x v_wmma_f32_16x16x32_bf16 (fp32 accumulate) ----
#pragma unroll
        for (int mi = 0; mi < 2; ++mi)
#pragma unroll
            for (int ni = 0; ni < 4; ++ni)
                c[mi][ni] = __builtin_amdgcn_wmma_f32_16x16x32_bf16(
                    false, a[mi], false, b[ni], (short)0, c[mi][ni], false, false);

        wait_asynccnt0();   // this wave's staged chunks landed in LDS
        __syncthreads();    // all waves' chunks visible; old buffer reusable
    }

    // ---- epilogue: bias + sigmoid(PHI*x), store bf16 ----
    // D layout (16x16 f32): VGPR v, lane l: m = 8*h + v, n = l&15
#pragma unroll
    for (int mi = 0; mi < 2; ++mi) {
#pragma unroll
        for (int ni = 0; ni < 4; ++ni) {
#pragma unroll
            for (int v = 0; v < 8; ++v) {
                int m = m0 + wm + 16 * mi + 8 * h + v;
                int n = n0 + wn + 16 * ni + r;
                float y = c[mi][ni][v] + bias[n];
                y = 1.0f / (1.0f + __expf(-PHI_F * y));
                out[(size_t)m * N + n] = (bf16_t)y;
            }
        }
    }
}

// ---------------------------------------------------------------------------
// Final layer: out[row] = dot(h2[row,:], W3) + b3   (N=1, fp32 output)
// One 256-thread block per row; fp32 accumulate, LDS tree reduction.
// ---------------------------------------------------------------------------
__global__ __launch_bounds__(256)
void final_dot(const bf16_t* __restrict__ H, const float* __restrict__ W3,
               const float* __restrict__ b3, float* __restrict__ out, int K) {
    __shared__ float red[256];
    const int row = blockIdx.x;
    const bf16_t* h = H + (size_t)row * K;
    float acc = 0.0f;
    for (int k = threadIdx.x; k < K; k += 256)
        acc += (float)h[k] * W3[k];
    red[threadIdx.x] = acc;
    __syncthreads();
    for (int s = 128; s > 0; s >>= 1) {
        if (threadIdx.x < s) red[threadIdx.x] += red[threadIdx.x + s];
        __syncthreads();
    }
    if (threadIdx.x == 0) out[row] = red[0] + b3[0];
}

// ---------------------------------------------------------------------------
extern "C" void kernel_launch(void* const* d_in, const int* in_sizes, int n_in,
                              void* d_out, int out_size, void* d_ws, size_t ws_size,
                              hipStream_t stream) {
    const float* x  = (const float*)d_in[0];
    const float* W1 = (const float*)d_in[1];
    const float* b1 = (const float*)d_in[2];
    const float* W2 = (const float*)d_in[3];
    const float* b2 = (const float*)d_in[4];
    const float* W3 = (const float*)d_in[5];
    const float* b3 = (const float*)d_in[6];
    float* out = (float*)d_out;

    // workspace carve (all sizes already 256B-aligned)
    char* ws = (char*)d_ws;
    size_t off = 0;
    bf16_t* xb  = (bf16_t*)(ws + off); off += (size_t)B_DIM * DIN * sizeof(bf16_t);
    bf16_t* w1b = (bf16_t*)(ws + off); off += (size_t)DH * DIN * sizeof(bf16_t);
    bf16_t* w2b = (bf16_t*)(ws + off); off += (size_t)DH * DH * sizeof(bf16_t);
    bf16_t* h1b = (bf16_t*)(ws + off); off += (size_t)B_DIM * DH * sizeof(bf16_t);
    bf16_t* h2b = (bf16_t*)(ws + off); off += (size_t)B_DIM * DH * sizeof(bf16_t);

    // fp32 -> bf16 staging
    cvt_f32_to_bf16<<<2048, 256, 0, stream>>>(x,  xb,  ((long)B_DIM * DIN) / 4);
    cvt_f32_to_bf16<<<2048, 256, 0, stream>>>(W1, w1b, ((long)DH * DIN) / 4);
    cvt_f32_to_bf16<<<2048, 256, 0, stream>>>(W2, w2b, ((long)DH * DH) / 4);

    // layer 1: [16384,2048] x [4096,2048]^T -> h1
    dim3 blk(256);
    dim3 g1(DH / 256, B_DIM / 64);
    mlp_gemm_bf16<<<g1, blk, 0, stream>>>(xb, w1b, b1, h1b, B_DIM, DH, DIN);
    // layer 2: [16384,4096] x [4096,4096]^T -> h2
    mlp_gemm_bf16<<<g1, blk, 0, stream>>>(h1b, w2b, b2, h2b, B_DIM, DH, DH);
    // layer 3: [16384,4096] x [4096,1] -> out
    final_dot<<<B_DIM, 256, 0, stream>>>(h2b, W3, b3, out, DH);
}
